// MyViT_13108240187938
// MI455X (gfx1250) — matrix operand order repaired and checked by reading, not compiled, not run on hardware
//
#include <hip/hip_runtime.h>
#include <math.h>
#include <stdint.h>

typedef __bf16 bf16_t;
typedef bf16_t v16bf __attribute__((ext_vector_type(16)));
typedef float  v8f   __attribute__((ext_vector_type(8)));

__device__ __forceinline__ uint16_t f2bf(float f) {
  union { float f; uint32_t u; } c; c.f = f;
  uint32_t u = c.u;
  return (uint16_t)((u + 0x7FFFu + ((u >> 16) & 1u)) >> 16);  // RNE f32->bf16
}
__device__ __forceinline__ uint32_t packbf2(float lo, float hi) {
  return (uint32_t)f2bf(lo) | ((uint32_t)f2bf(hi) << 16);
}

union FragAB { v16bf v; uint32_t u[8]; };
union FragC  { v8f   v; float    f[8]; };

#define EPI_NONE   0
#define EPI_POSEMB 1
#define EPI_SCALE  2
#define EPI_GELU   3
#define EPI_RESID  4

// C = A(MxK,f32) * B(KxN,f32) [+bias][epilogue], bf16 WMMA, f32 accumulation.
// Block tile: 64(M) x 128(N) x 32(K); 256 threads = 8 waves in a 2x4 grid;
// each wave computes a 32x32 macro-tile = 2x2 v_wmma_f32_16x16x32_bf16 per K-step
// (A/B fragments each reused twice). Batched over blockIdx.z via element strides.
// TRANSB: B element (k,n) at B[n*ldb+k] (for Q.K^T).
template<bool TRANSB, int EPI>
__global__ __launch_bounds__(256)
void gemm_wmma_bf16(const float* __restrict__ A, const float* __restrict__ B,
                    const float* __restrict__ bias, const float* __restrict__ resid,
                    float* __restrict__ C,
                    int M, int N, int K, int lda, int ldb, int ldc,
                    long aBatch, long bBatch, long cBatch, long biasBatch,
                    float scale)
{
  constexpr int ASTR = 36, BSTR = 36;           // even pad -> 4B-aligned bf16 pairs
  __shared__ uint16_t sA[64 * ASTR];            // A tile, [m][k] bf16
  __shared__ uint16_t sB[128 * BSTR];           // B tile transposed, [n][k] bf16

  const int z = blockIdx.z;
  const float* Ag = A + (size_t)z * aBatch;
  const float* Bg = B + (size_t)z * bBatch;
  float*       Cg = C + (size_t)z * cBatch;
  const float* biasg  = bias  ? bias  + (size_t)z * biasBatch : nullptr;
  const float* residg = resid ? resid + (size_t)z * cBatch    : nullptr;

  const int tid  = threadIdx.x;
  const int lane = tid & 31;
  const int w    = tid >> 5;       // 0..7
  const int wm   = w >> 2;         // 0..1  (M macro-tile)
  const int wn   = w & 3;          // 0..3  (N macro-tile)
  const int lrow = lane & 15;
  const int lhi  = lane >> 4;

  const int row0 = blockIdx.y * 64;
  const int col0 = blockIdx.x * 128;

  FragC acc[2][2];
  #pragma unroll
  for (int i = 0; i < 2; ++i)
    #pragma unroll
    for (int j = 0; j < 2; ++j)
      #pragma unroll
      for (int e = 0; e < 8; ++e) acc[i][j].f[e] = 0.0f;

  for (int kk = 0; kk < K; kk += 32) {
    // ---- stage A 64x32 (f32 -> bf16), 2x float4 per thread, packed b32 stores ----
    #pragma unroll
    for (int it = 0; it < 2; ++it) {
      int idx = tid + it * 256;
      int r   = idx >> 3;
      int k4  = (idx & 7) << 2;
      const float4 av = *(const float4*)(Ag + (size_t)(row0 + r) * lda + kk + k4);
      *(uint32_t*)&sA[r * ASTR + k4]     = packbf2(av.x, av.y);
      *(uint32_t*)&sA[r * ASTR + k4 + 2] = packbf2(av.z, av.w);
    }
    // ---- stage B 32x128 transposed into sB[n][k] ----
    if (TRANSB) {
      #pragma unroll
      for (int it = 0; it < 4; ++it) {
        int idx = tid + it * 256;
        int n   = idx >> 3;
        int k4  = (idx & 7) << 2;
        const float4 bv = *(const float4*)(Bg + (size_t)(col0 + n) * ldb + kk + k4);
        *(uint32_t*)&sB[n * BSTR + k4]     = packbf2(bv.x, bv.y);
        *(uint32_t*)&sB[n * BSTR + k4 + 2] = packbf2(bv.z, bv.w);
      }
    } else {
      #pragma unroll
      for (int it = 0; it < 4; ++it) {
        int idx = tid + it * 256;
        int k   = idx >> 5;
        int n4  = (idx & 31) << 2;
        const float4 bv = *(const float4*)(Bg + (size_t)(kk + k) * ldb + col0 + n4);
        sB[(n4 + 0) * BSTR + k] = f2bf(bv.x);
        sB[(n4 + 1) * BSTR + k] = f2bf(bv.y);
        sB[(n4 + 2) * BSTR + k] = f2bf(bv.z);
        sB[(n4 + 3) * BSTR + k] = f2bf(bv.w);
      }
    }
    __syncthreads();

    // ---- fragments per ISA 7.12.2 layouts; 2x2 WMMAs with fragment reuse ----
    FragAB fa[2], fb[2];
    #pragma unroll
    for (int i = 0; i < 2; ++i) {
      #pragma unroll
      for (int v = 0; v < 8; ++v) {
        // A 16x32 bf16: lane<16 K={2v(+16 if v>=4)}, lane>=16 adds +8
        int ka = ((v & 4) << 2) + ((v & 3) << 1) + (lhi << 3);
        fa[i].u[v] = *(const uint32_t*)&sA[(wm * 32 + i * 16 + lrow) * ASTR + ka];
        // B 32x16 bf16: same K order as A: lane<16 K={2v(+16 if v>=4)}, lane>=16 adds +8
        int kb = ((v & 4) << 2) + ((v & 3) << 1) + (lhi << 3);
        fb[i].u[v] = *(const uint32_t*)&sB[(wn * 32 + i * 16 + lrow) * BSTR + kb];
      }
    }
    #pragma unroll
    for (int i = 0; i < 2; ++i)
      #pragma unroll
      for (int j = 0; j < 2; ++j)
        acc[i][j].v = __builtin_amdgcn_wmma_f32_16x16x32_bf16(
            false, fa[i].v, false, fb[j].v, (short)0, acc[i][j].v, false, false);
    __syncthreads();
  }

  // ---- epilogue: C/D layout m = v + 8*lhi, n = lane&15 ----
  #pragma unroll
  for (int j = 0; j < 2; ++j) {
    const int n = col0 + wn * 32 + j * 16 + lrow;
    const float bval = biasg ? biasg[n] : 0.0f;
    #pragma unroll
    for (int i = 0; i < 2; ++i) {
      const int mbase = row0 + wm * 32 + i * 16 + lhi * 8;
      #pragma unroll
      for (int v = 0; v < 8; ++v) {
        int m = mbase + v;
        float x = acc[i][j].f[v] + bval;
        if (EPI == EPI_SCALE)  x = acc[i][j].f[v] * scale;
        if (EPI == EPI_POSEMB) {
          float jj = (float)(n & ~1);
          float ang = (float)m * __expf(jj * (-9.210340371976184f / 2048.0f)); // 10000^(-j/D)
          x += (n & 1) ? __cosf(ang) : __sinf(ang);
        }
        if (EPI == EPI_GELU)   x = 0.5f * x * (1.0f + erff(x * 0.7071067811865476f));
        if (EPI == EPI_RESID)  x += residg[(size_t)m * ldc + n];
        Cg[(size_t)m * ldc + n] = x;
      }
    }
  }
}

__global__ __launch_bounds__(256)
void layernorm2048(const float* __restrict__ x, const float* __restrict__ g,
                   const float* __restrict__ b, float* __restrict__ y)
{
  __shared__ float rs[256], rs2[256];
  const int row = blockIdx.x, tid = threadIdx.x;
  const float* xr = x + (size_t)row * 2048;
  float s = 0.f, s2 = 0.f;
  #pragma unroll
  for (int i = 0; i < 8; ++i) { float v = xr[tid + i * 256]; s += v; s2 += v * v; }
  rs[tid] = s; rs2[tid] = s2; __syncthreads();
  for (int off = 128; off > 0; off >>= 1) {
    if (tid < off) { rs[tid] += rs[tid + off]; rs2[tid] += rs2[tid + off]; }
    __syncthreads();
  }
  const float mean = rs[0] * (1.0f / 2048.0f);
  const float var  = rs2[0] * (1.0f / 2048.0f) - mean * mean;
  const float rstd = rsqrtf(var + 1e-5f);
  float* yr = y + (size_t)row * 2048;
  #pragma unroll
  for (int i = 0; i < 8; ++i) {
    int c = tid + i * 256;
    yr[c] = (xr[c] - mean) * rstd * g[c] + b[c];
  }
}

__global__ __launch_bounds__(256)
void softmax256(float* __restrict__ att)
{
  __shared__ float red[256];
  const int tid = threadIdx.x;
  float* p = att + (size_t)blockIdx.x * 256;
  const float v = p[tid];
  red[tid] = v; __syncthreads();
  for (int off = 128; off > 0; off >>= 1) {
    if (tid < off) red[tid] = fmaxf(red[tid], red[tid + off]);
    __syncthreads();
  }
  const float mx = red[0]; __syncthreads();
  const float e = __expf(v - mx);
  red[tid] = e; __syncthreads();
  for (int off = 128; off > 0; off >>= 1) {
    if (tid < off) red[tid] += red[tid + off];
    __syncthreads();
  }
  p[tid] = e / red[0];
}

__global__ __launch_bounds__(256)
void head_kernel(const float* __restrict__ x, const float* __restrict__ hw,
                 const float* __restrict__ hb, const float* __restrict__ mem,
                 float* __restrict__ out)
{
  __shared__ float red[256];
  const int tid = threadIdx.x;
  float s = 0.f;
  for (int i = tid; i < 2048; i += 256) s += x[i] * hw[i];
  red[tid] = s; __syncthreads();
  for (int off = 128; off > 0; off >>= 1) {
    if (tid < off) red[tid] += red[tid + off];
    __syncthreads();
  }
  if (tid == 0) out[0] = 1.0f / (1.0f + __expf(-(red[0] + hb[0])));
  if (tid < 8)  out[1 + tid] = mem[tid];   // memory passthrough (output #2)
}

extern "C" void kernel_launch(void* const* d_in, const int* in_sizes, int n_in,
                              void* d_out, int out_size, void* d_ws, size_t ws_size,
                              hipStream_t stream)
{
  (void)in_sizes; (void)n_in; (void)out_size; (void)ws_size;
  const float* images = (const float*)d_in[0];
  const float* memory = (const float*)d_in[1];
  const float* wmap   = (const float*)d_in[2];
  const float* bmap   = (const float*)d_in[3];
  const float* ln1_g  = (const float*)d_in[4];
  const float* ln1_b  = (const float*)d_in[5];
  const float* qw     = (const float*)d_in[6];
  const float* qb     = (const float*)d_in[7];
  const float* kw     = (const float*)d_in[8];
  const float* kb     = (const float*)d_in[9];
  const float* vw     = (const float*)d_in[10];
  const float* vb     = (const float*)d_in[11];
  const float* ln2_g  = (const float*)d_in[12];
  const float* ln2_b  = (const float*)d_in[13];
  const float* w1     = (const float*)d_in[14];
  const float* b1     = (const float*)d_in[15];
  const float* w2     = (const float*)d_in[16];
  const float* b2     = (const float*)d_in[17];
  const float* head_w = (const float*)d_in[18];
  const float* head_b = (const float*)d_in[19];

  float* ws  = (float*)d_ws;
  float* x   = ws;                       // 256*2048
  float* h   = x + 256 * 2048;           // 256*2048
  float* q   = h + 256 * 2048;           // 256*2048
  float* k   = q + 256 * 2048;           // 256*2048
  float* v   = k + 256 * 2048;           // 256*2048
  float* att = v + 256 * 2048;           // 16*256*256
  float* t   = att + 16 * 256 * 256;     // 256*8192

  const dim3 blk(256);
  const float inv_scale = 1.0f / sqrtf(128.0f);

  // x = images(256x4096) @ wmap + bmap + posemb
  gemm_wmma_bf16<false, EPI_POSEMB><<<dim3(2048 / 128, 256 / 64, 1), blk, 0, stream>>>(
      images, wmap, bmap, nullptr, x, 256, 2048, 4096, 4096, 2048, 2048,
      0, 0, 0, 0, 0.0f);

  for (int l = 0; l < 2; ++l) {
    const float* qwl = qw + (size_t)l * 16 * 128 * 128;
    const float* kwl = kw + (size_t)l * 16 * 128 * 128;
    const float* vwl = vw + (size_t)l * 16 * 128 * 128;
    const float* qbl = qb + (size_t)l * 16 * 128;
    const float* kbl = kb + (size_t)l * 16 * 128;
    const float* vbl = vb + (size_t)l * 16 * 128;

    layernorm2048<<<256, blk, 0, stream>>>(x, ln1_g + l * 2048, ln1_b + l * 2048, h);

    // per-head q/k/v: (256x128)@(128x128), batched z=16 via column offsets
    gemm_wmma_bf16<false, EPI_NONE><<<dim3(1, 4, 16), blk, 0, stream>>>(
        h, qwl, qbl, nullptr, q, 256, 128, 128, 2048, 128, 2048,
        128, 128 * 128, 128, 128, 0.0f);
    gemm_wmma_bf16<false, EPI_NONE><<<dim3(1, 4, 16), blk, 0, stream>>>(
        h, kwl, kbl, nullptr, k, 256, 128, 128, 2048, 128, 2048,
        128, 128 * 128, 128, 128, 0.0f);
    gemm_wmma_bf16<false, EPI_NONE><<<dim3(1, 4, 16), blk, 0, stream>>>(
        h, vwl, vbl, nullptr, v, 256, 128, 128, 2048, 128, 2048,
        128, 128 * 128, 128, 128, 0.0f);

    // att = (q @ k^T) / sqrt(dh), per head
    gemm_wmma_bf16<true, EPI_SCALE><<<dim3(256 / 128, 4, 16), blk, 0, stream>>>(
        q, k, nullptr, nullptr, att, 256, 256, 128, 2048, 2048, 256,
        128, 128, (long)256 * 256, 0, inv_scale);

    softmax256<<<16 * 256, blk, 0, stream>>>(att);

    // x = x + att @ v, per head
    gemm_wmma_bf16<false, EPI_RESID><<<dim3(1, 4, 16), blk, 0, stream>>>(
        att, v, nullptr, x, x, 256, 128, 256, 256, 2048, 2048,
        (long)256 * 256, 128, 128, 0, 0.0f);

    layernorm2048<<<256, blk, 0, stream>>>(x, ln2_g + l * 2048, ln2_b + l * 2048, h);

    // t = gelu(h @ w1 + b1)
    gemm_wmma_bf16<false, EPI_GELU><<<dim3(8192 / 128, 4, 1), blk, 0, stream>>>(
        h, w1 + (size_t)l * 2048 * 8192, b1 + (size_t)l * 8192, nullptr, t,
        256, 8192, 2048, 2048, 8192, 8192, 0, 0, 0, 0, 0.0f);

    // x = x + t @ w2 + b2
    gemm_wmma_bf16<false, EPI_RESID><<<dim3(2048 / 128, 4, 1), blk, 0, stream>>>(
        t, w2 + (size_t)l * 8192 * 2048, b2 + (size_t)l * 2048, x, x,
        256, 2048, 8192, 8192, 2048, 2048, 0, 0, 0, 0, 0.0f);
  }

  head_kernel<<<1, blk, 0, stream>>>(x, head_w, head_b, memory, (float*)d_out);
}
